// PrototypicalSpanNER_34394098106376
// MI455X (gfx1250) — matrix operand order repaired and checked by reading, not compiled, not run on hardware
//
#include <hip/hip_runtime.h>
#include <hip/hip_bf16.h>
#include <math.h>

// ---------- problem constants (fixed by the reference) ----------
#define T_    65536
#define H_    768
#define K3    2304        // 3*H
#define NSUP  32768
#define NQRY  32768
#define MTOT  65536       // NSUP + NQRY
#define NCLS  10

typedef __attribute__((ext_vector_type(16))) __bf16    v16bf;
typedef __attribute__((ext_vector_type(8)))  __bf16    v8bf;
typedef __attribute__((ext_vector_type(8)))  float     v8f;
typedef __attribute__((ext_vector_type(4)))  uint32_t  v4u;
typedef __attribute__((ext_vector_type(8)))  uint32_t  v8u;

// =====================================================================
// TDM descriptor build + issue (2-D tile, 2-group form; VADDR2/3 = NULL)
//   elem = bf16 (data_size code 1 -> 2 bytes)
//   tensor: rows x cols (row-major, stride = row_stride elements)
//   tile:   tdim1 rows x tdim0 cols, loaded to lds_off
// =====================================================================
__device__ __forceinline__ void tdm_load_2d(uint64_t gaddr, uint32_t lds_off,
                                            uint32_t cols, uint32_t rows,
                                            uint32_t tdim0, uint32_t tdim1,
                                            uint32_t row_stride) {
    v4u g0;
    g0.x = 1u;                                       // count=1, user desc
    g0.y = lds_off;                                  // LDS byte address
    g0.z = (uint32_t)gaddr;                          // global_addr[31:0]
    g0.w = (uint32_t)(gaddr >> 32) | (2u << 30);     // global_addr[56:32] | type=2

    v8u g1;
    g1.s0 = (1u << 16);                              // wg_mask=0, data_size=1 (2B)
    g1.s1 = (cols & 0xffffu) << 16;                  // tensor_dim0[15:0] @ bits 63:48
    g1.s2 = (cols >> 16) | ((rows & 0xffffu) << 16); // td0[31:16] | td1[15:0]
    g1.s3 = (rows >> 16) | (tdim0 << 16);            // td1[31:16] | tile_dim0
    g1.s4 = tdim1;                                   // tile_dim1 | tile_dim2=0
    g1.s5 = row_stride;                              // tensor_dim0_stride[31:0]
    g1.s6 = 0u;                                      // stride0[47:32] | stride1[15:0]
    g1.s7 = 0u;                                      // stride1[47:16]

    asm volatile("tensor_load_to_lds %0, %1" :: "s"(g0), "s"(g1) : "memory");
}

// =====================================================================
// 0) W_proj (2304x768 f32) -> bf16
// =====================================================================
__global__ void k_cvt_w(const float* __restrict__ W, __bf16* __restrict__ Wbf) {
    int i = blockIdx.x * 256 + threadIdx.x;
    if (i < K3 * H_) Wbf[i] = (__bf16)W[i];
}

// =====================================================================
// 1) span features: [emb[s], emb[e-1], mean(emb[s:e])] -> bf16 (row = 2304)
// =====================================================================
__global__ void k_feat(const float* __restrict__ emb,
                       const int*   __restrict__ spans,
                       __bf16*      __restrict__ feat,
                       int nspans) {
    int i = blockIdx.x;
    if (i >= nspans) return;
    int s = spans[2 * i];
    int e = spans[2 * i + 1];
    int len = e - s;
    float inv = 1.0f / (float)len;
    const float* rs = emb + (size_t)s * H_;
    const float* re = emb + (size_t)(e - 1) * H_;
    __bf16* out = feat + (size_t)i * K3;
    for (int d = threadIdx.x; d < H_; d += 256) {
        float sum = 0.0f;
        for (int t = 0; t < len; ++t)
            sum += emb[(size_t)(s + t) * H_ + d];
        out[d]          = (__bf16)rs[d];
        out[H_ + d]     = (__bf16)re[d];
        out[2 * H_ + d] = (__bf16)(sum * inv);
    }
}

// =====================================================================
// 2) GEMM + bias + exact GELU:  emb(M x 768) = gelu(feat @ Wbf + b)
//    block tile M=64 x N=128; 8 waves = 4(M) x 2(N); each wave: 16x64
//    -> 4 x v_wmma_f32_16x16x32_bf16 per K-step (A-frag reused 4x)
//    A (64x32) and B (32x128) tiles staged in LDS by the Tensor Data Mover
// =====================================================================
__global__ __launch_bounds__(256)
void k_gemm_gelu(const __bf16* __restrict__ feat,
                 const __bf16* __restrict__ Wbf,
                 const float*  __restrict__ bias,
                 __bf16*       __restrict__ embout) {
    __shared__ __bf16 As[64 * 32];    // 4 KB
    __shared__ __bf16 Bs[32 * 128];   // 8 KB

    const int tid   = threadIdx.x;
    const int lane  = tid & 31;
    const int wave  = tid >> 5;          // 0..7
    const int mwave = (wave & 3) * 16;   // 0,16,32,48
    const int nwave = (wave >> 2) * 64;  // 0,64
    const int m0    = blockIdx.x * 64;
    const int n0b   = blockIdx.y * 128;

    v8f acc0 = {}, acc1 = {}, acc2 = {}, acc3 = {};

    const uint32_t ldsA = (uint32_t)(uintptr_t)&As[0];
    const uint32_t ldsB = (uint32_t)(uintptr_t)&Bs[0];

    for (int k0 = 0; k0 < K3; k0 += 32) {
        __syncthreads();               // previous fragments consumed
        if (wave == 0) {
            // A tile: 64 rows x 32 cols from feat (stride K3)
            tdm_load_2d((uint64_t)(uintptr_t)(feat + (size_t)m0 * K3 + k0),
                        ldsA, (uint32_t)K3, (uint32_t)MTOT, 32u, 64u, (uint32_t)K3);
            // B tile: 32 rows x 128 cols from Wbf (stride H_)
            tdm_load_2d((uint64_t)(uintptr_t)(Wbf + (size_t)k0 * H_ + n0b),
                        ldsB, (uint32_t)H_, (uint32_t)K3, 128u, 32u, (uint32_t)H_);
            __builtin_amdgcn_s_wait_tensorcnt(0);
        }
        // prefetch next K tiles toward GL2 for the TDM (global_prefetch_b8)
        if (k0 + 32 < K3) {
            __builtin_prefetch(&feat[(size_t)(m0 + (tid >> 2)) * K3 + k0 + 32 + (tid & 3) * 8], 0, 2);
            __builtin_prefetch(&Wbf[(size_t)(k0 + 32 + (tid >> 3)) * H_ + n0b + (tid & 7) * 16], 0, 2);
        }
        __syncthreads();               // tiles visible to all waves

        // ---- fragments per ISA 16-bit A (16x32) / B (32x16) layouts ----
        const int arow = mwave + (lane & 15);
        const int kb   = (lane >> 4) * 8;
        v8bf alo = *(const v8bf*)&As[arow * 32 + kb];
        v8bf ahi = *(const v8bf*)&As[arow * 32 + kb + 16];
        v16bf a = __builtin_shufflevector(alo, ahi,
                    0, 1, 2, 3, 4, 5, 6, 7, 8, 9, 10, 11, 12, 13, 14, 15);

        const __bf16* brow = &Bs[nwave + (lane & 15)];
#define B_FRAG(J)                                                           \
        ({ v16bf _b;                                                        \
           for (int _s = 0; _s < 16; ++_s)                                  \
               _b[_s] = brow[(J) * 16 + (kb + (_s & 7) + (_s >> 3) * 16) * 128]; \
           _b; })

        v16bf b0 = B_FRAG(0), b1 = B_FRAG(1), b2 = B_FRAG(2), b3 = B_FRAG(3);
        acc0 = __builtin_amdgcn_wmma_f32_16x16x32_bf16(false, a, false, b0, (short)0, acc0, false, false);
        acc1 = __builtin_amdgcn_wmma_f32_16x16x32_bf16(false, a, false, b1, (short)0, acc1, false, false);
        acc2 = __builtin_amdgcn_wmma_f32_16x16x32_bf16(false, a, false, b2, (short)0, acc2, false, false);
        acc3 = __builtin_amdgcn_wmma_f32_16x16x32_bf16(false, a, false, b3, (short)0, acc3, false, false);
#undef B_FRAG
    }

    // ---- epilogue: bias + exact GELU, store bf16 ----
    // C/D layout: VGPR i -> M = i + 8*(lane>=16), N = lane&15
    const int mb = m0 + mwave + (lane >> 4) * 8;
    v8f accs[4] = {acc0, acc1, acc2, acc3};
#pragma unroll
    for (int j = 0; j < 4; ++j) {
        const int nc = n0b + nwave + j * 16 + (lane & 15);
        const float bv = bias[nc];
#pragma unroll
        for (int i = 0; i < 8; ++i) {
            float x = accs[j][i] + bv;
            float g = 0.5f * x * (1.0f + erff(x * 0.70710678118654752f));
            embout[(size_t)(mb + i) * H_ + nc] = (__bf16)g;
        }
    }
}

// =====================================================================
// 3a) zero accumulators
// =====================================================================
__global__ void k_zero(float* __restrict__ p, int n) {
    int i = blockIdx.x * 256 + threadIdx.x;
    if (i < n) p[i] = 0.0f;
}

// =====================================================================
// 3b) segment-sum support embeddings into class sums + counts
// =====================================================================
__global__ __launch_bounds__(256)
void k_proto_acc(const __bf16* __restrict__ supemb,
                 const int*    __restrict__ labels,
                 float* __restrict__ sums,
                 float* __restrict__ counts) {
    __shared__ float lac[NCLS * H_];
    __shared__ float lcnt[NCLS];
    for (int i = threadIdx.x; i < NCLS * H_; i += 256) lac[i] = 0.0f;
    if (threadIdx.x < NCLS) lcnt[threadIdx.x] = 0.0f;
    __syncthreads();

    const int r0 = blockIdx.x * 128;
    for (int r = 0; r < 128; ++r) {
        const int rr  = r0 + r;
        const int lab = labels[rr];
        const __bf16* row = supemb + (size_t)rr * H_;
        for (int d = threadIdx.x; d < H_; d += 256)
            lac[lab * H_ + d] += (float)row[d];
        if (threadIdx.x == 0) lcnt[lab] += 1.0f;
    }
    __syncthreads();
    for (int i = threadIdx.x; i < NCLS * H_; i += 256)
        atomicAdd(&sums[i], lac[i]);
    if (threadIdx.x < NCLS)
        atomicAdd(&counts[threadIdx.x], lcnt[threadIdx.x]);
}

// =====================================================================
// 3c) finalize prototypes (gmean fallback) + L2-normalize (1 block)
// =====================================================================
__global__ __launch_bounds__(256)
void k_proto_fin(const float* __restrict__ sums,
                 const float* __restrict__ counts,
                 float* __restrict__ protosN) {
    __shared__ float protos[NCLS * H_];
    __shared__ float red[256];

    for (int d = threadIdx.x; d < H_; d += 256) {
        float tot = 0.0f;
#pragma unroll
        for (int c = 0; c < NCLS; ++c) tot += sums[c * H_ + d];
        float gm = tot / (float)NSUP;
#pragma unroll
        for (int c = 0; c < NCLS; ++c) {
            float cnt = counts[c];
            protos[c * H_ + d] = (cnt > 0.0f) ? sums[c * H_ + d] / fmaxf(cnt, 1.0f) : gm;
        }
    }
    __syncthreads();

    for (int c = 0; c < NCLS; ++c) {
        float p = 0.0f;
        for (int d = threadIdx.x; d < H_; d += 256) {
            float v = protos[c * H_ + d];
            p += v * v;
        }
        red[threadIdx.x] = p;
        __syncthreads();
        for (int s = 128; s > 0; s >>= 1) {
            if (threadIdx.x < s) red[threadIdx.x] += red[threadIdx.x + s];
            __syncthreads();
        }
        float inv = 1.0f / fmaxf(sqrtf(red[0]), 1e-12f);
        __syncthreads();
        for (int d = threadIdx.x; d < H_; d += 256)
            protosN[c * H_ + d] = protos[c * H_ + d] * inv;
        __syncthreads();
    }
}

// =====================================================================
// 4) cosine similarity + none-logit: one wave32 per query row
// =====================================================================
__global__ __launch_bounds__(256)
void k_sim(const __bf16* __restrict__ qryemb,
           const float*  __restrict__ protosN,
           float*        __restrict__ out) {
    __shared__ float P[NCLS * H_];
    for (int i = threadIdx.x; i < NCLS * H_; i += 256) P[i] = protosN[i];
    __syncthreads();

    const int lane = threadIdx.x & 31;
    const int wave = threadIdx.x >> 5;
    const int q    = blockIdx.x * 8 + wave;
    const __bf16* row = qryemb + (size_t)q * H_;

    float dot[NCLS];
#pragma unroll
    for (int c = 0; c < NCLS; ++c) dot[c] = 0.0f;
    float qq = 0.0f;

    for (int d = lane; d < H_; d += 32) {
        float v = (float)row[d];
        qq += v * v;
#pragma unroll
        for (int c = 0; c < NCLS; ++c) dot[c] += v * P[c * H_ + d];
    }
#pragma unroll
    for (int off = 16; off > 0; off >>= 1) {
        qq += __shfl_xor(qq, off, 32);
#pragma unroll
        for (int c = 0; c < NCLS; ++c) dot[c] += __shfl_xor(dot[c], off, 32);
    }
    if (lane == 0) {
        float inv = 1.0f / fmaxf(sqrtf(qq), 1e-12f);
#pragma unroll
        for (int c = 0; c < NCLS; ++c) out[(size_t)q * (NCLS + 1) + c] = dot[c] * inv;
        out[(size_t)q * (NCLS + 1) + NCLS] = 0.5f;
    }
}

// =====================================================================
// launcher
// =====================================================================
extern "C" void kernel_launch(void* const* d_in, const int* in_sizes, int n_in,
                              void* d_out, int out_size, void* d_ws, size_t ws_size,
                              hipStream_t stream) {
    const float* token_emb = (const float*)d_in[0];
    const int*   spans_sup = (const int*)d_in[1];
    const int*   labels    = (const int*)d_in[2];
    const int*   spans_qry = (const int*)d_in[3];
    const float* W_proj    = (const float*)d_in[4];
    const float* b_proj    = (const float*)d_in[5];
    float*       out       = (float*)d_out;

    char* ws = (char*)d_ws;
    size_t off = 0;
    __bf16* Wbf    = (__bf16*)(ws + off); off += (size_t)K3 * H_ * 2;
    __bf16* featbf = (__bf16*)(ws + off); off += (size_t)MTOT * K3 * 2;
    __bf16* embbf  = (__bf16*)(ws + off); off += (size_t)MTOT * H_ * 2;
    float*  sums   = (float*)(ws + off);  off += (size_t)NCLS * H_ * 4;
    float*  counts = (float*)(ws + off);  off += 256;
    float*  protosN= (float*)(ws + off);  off += (size_t)NCLS * H_ * 4;

    k_cvt_w<<<(K3 * H_ + 255) / 256, 256, 0, stream>>>(W_proj, Wbf);

    k_feat<<<NSUP, 256, 0, stream>>>(token_emb, spans_sup, featbf, NSUP);
    k_feat<<<NQRY, 256, 0, stream>>>(token_emb, spans_qry,
                                     featbf + (size_t)NSUP * K3, NQRY);

    dim3 ggrid(MTOT / 64, H_ / 128);
    k_gemm_gelu<<<ggrid, 256, 0, stream>>>(featbf, Wbf, b_proj, embbf);

    k_zero<<<(NCLS * H_ + NCLS + 255) / 256, 256, 0, stream>>>(sums, NCLS * H_ + NCLS);
    k_proto_acc<<<NSUP / 128, 256, 0, stream>>>(embbf, labels, sums, counts);
    k_proto_fin<<<1, 256, 0, stream>>>(sums, counts, protosN);

    k_sim<<<NQRY / 8, 256, 0, stream>>>(embbf + (size_t)NSUP * H_, protosN, out);
}
